// FlashAttention_82514911690912
// MI455X (gfx1250) — matrix operand order repaired and checked by reading, not compiled, not run on hardware
//
#include <hip/hip_runtime.h>

// ---------------------------------------------------------------------------
// CDNA5 (gfx1250) multi-head causal attention forward, bf16 WMMA pipeline
// with async global->LDS double-buffered K/V staging in the attention kernel.
// ---------------------------------------------------------------------------

typedef __attribute__((ext_vector_type(16))) __bf16 v16bf;
typedef __attribute__((ext_vector_type(8)))  __bf16 v8bf;
typedef __attribute__((ext_vector_type(4)))  __bf16 v4bf;
typedef __attribute__((ext_vector_type(8)))  float  v8f;

#define BATCH 2
#define SEQ   2048
#define EMB   2048
#define HEADS 16
#define HDIM  128

union FragU { v16bf v; struct { v8bf lo, hi; } h; };

// A fragment (16x32 bf16) from row-major storage, leading dim `ld` (elements).
// ISA layout: lane L: row m = L%16; k = 8*(L/16)+j (elems 0..7), 16+8*(L/16)+j (elems 8..15)
__device__ inline v16bf load_frag_a(const __bf16* base, int ld) {
    const int lane = threadIdx.x & 31;
    const int m = lane & 15, hi = lane >> 4;
    const __bf16* p = base + m * ld + 8 * hi;
    FragU f;
    f.h.lo = *(const v8bf*)(p);
    f.h.hi = *(const v8bf*)(p + 16);
    return f.v;
}

// B fragment (32x16 bf16) from n-major storage: element (k,n) at base[n*ld+k].
// ISA layout: lane L: col n = L%16; k = 8*(L/16)+j (elems 0..7), 16+8*(L/16)+j (elems 8..15)
__device__ inline v16bf load_frag_b(const __bf16* base, int ld) {
    const int lane = threadIdx.x & 31;
    const int n = lane & 15, hi = lane >> 4;
    const __bf16* p = base + n * ld + 8 * hi;
    FragU f;
    f.h.lo = *(const v8bf*)(p);
    f.h.hi = *(const v8bf*)(p + 16);
    return f.v;
}

__device__ inline v8f wmma_bf16(v16bf a, v16bf b, v8f c) {
    return __builtin_amdgcn_wmma_f32_16x16x32_bf16(false, a, false, b,
                                                   (short)0, c, false, false);
}

// Async global -> LDS copy, 16 bytes per lane. Tracked by ASYNCcnt.
// VDST VGPR carries the wave-relative LDS byte offset (= low 32 bits of the
// generic pointer, per the LDS aperture mapping addr[31:0] -> LDS offset).
__device__ inline void async_copy_b128(unsigned lds_off, const void* gptr) {
    asm volatile("global_load_async_to_lds_b128 %0, %1, off"
                 :: "v"(lds_off), "v"(gptr) : "memory");
}

// ---------------------------------------------------------------------------
// GEMM: C[MxN] = A[MxK](f32) * B[KxN](f32), bf16 WMMA compute, f32 accumulate.
// Block tile 128x128, BK=32. 8 waves: wave w -> rows (w/2)*32, cols (w%2)*64.
// ---------------------------------------------------------------------------
template <bool F32OUT>
__global__ __launch_bounds__(256) void gemm_f32_bf16(
    const float* __restrict__ A, const float* __restrict__ Bw,
    void* __restrict__ Cout, int M, int N, int K)
{
    __shared__ __bf16 As[128 * 40];   // row-major 128x32 (+8 pad)
    __shared__ __bf16 Bt[128 * 40];   // n-major   128x32 (+8 pad)

    const int t = threadIdx.x;
    const int m0 = blockIdx.y * 128;
    const int n0 = blockIdx.x * 128;
    const int w = t >> 5;
    const int waveM = (w >> 1) * 32;
    const int waveN = (w & 1) * 64;

    v8f acc[2][4];
    for (int i = 0; i < 2; ++i)
        for (int j = 0; j < 4; ++j)
            acc[i][j] = v8f{0.f, 0.f, 0.f, 0.f, 0.f, 0.f, 0.f, 0.f};

    for (int k0 = 0; k0 < K; k0 += 32) {
        // Stage A tile (convert f32 -> bf16)
        #pragma unroll
        for (int i = 0; i < 4; ++i) {
            const int id  = i * 256 + t;      // 1024 float4 loads
            const int row = id >> 3;          // 8 float4 per 32-float row
            const int c4  = id & 7;
            const float4 f =
                *(const float4*)(A + (size_t)(m0 + row) * K + k0 + c4 * 4);
            v4bf o;
            o[0] = (__bf16)f.x; o[1] = (__bf16)f.y;
            o[2] = (__bf16)f.z; o[3] = (__bf16)f.w;
            *(v4bf*)(&As[row * 40 + c4 * 4]) = o;
        }
        // Stage B tile transposed (n-major) so B-fragments are contiguous
        #pragma unroll
        for (int i = 0; i < 4; ++i) {
            const int id = i * 256 + t;
            const int kk = id >> 5;           // 32 float4 per 128-float row
            const int c4 = id & 31;
            const float4 f =
                *(const float4*)(Bw + (size_t)(k0 + kk) * N + n0 + c4 * 4);
            Bt[(c4 * 4 + 0) * 40 + kk] = (__bf16)f.x;
            Bt[(c4 * 4 + 1) * 40 + kk] = (__bf16)f.y;
            Bt[(c4 * 4 + 2) * 40 + kk] = (__bf16)f.z;
            Bt[(c4 * 4 + 3) * 40 + kk] = (__bf16)f.w;
        }
        __syncthreads();

        v16bf af[2];
        #pragma unroll
        for (int cm = 0; cm < 2; ++cm)
            af[cm] = load_frag_a(&As[(waveM + cm * 16) * 40], 40);
        #pragma unroll
        for (int cn = 0; cn < 4; ++cn) {
            v16bf bf = load_frag_b(&Bt[(waveN + cn * 16) * 40], 40);
            #pragma unroll
            for (int cm = 0; cm < 2; ++cm)
                acc[cm][cn] = wmma_bf16(af[cm], bf, acc[cm][cn]);
        }
        __syncthreads();
    }

    // Epilogue: C layout lane L -> n = L%16, m = v + 8*(L/16)
    const int lane = t & 31;
    const int cN = lane & 15, hiL = lane >> 4;
    #pragma unroll
    for (int cm = 0; cm < 2; ++cm)
        #pragma unroll
        for (int cn = 0; cn < 4; ++cn)
            #pragma unroll
            for (int v = 0; v < 8; ++v) {
                const int m = m0 + waveM + cm * 16 + v + 8 * hiL;
                const int n = n0 + waveN + cn * 16 + cN;
                if (F32OUT)
                    ((float*)Cout)[(size_t)m * N + n] = acc[cm][cn][v];
                else
                    ((__bf16*)Cout)[(size_t)m * N + n] = (__bf16)acc[cm][cn][v];
            }
}

// ---------------------------------------------------------------------------
// Reorder qkv[b*s, 6144] -> Q[b,h,s,d], K[b,h,s,d], Vt[b,h,d,s]
// ---------------------------------------------------------------------------
__global__ __launch_bounds__(256) void reorder_qkv(
    const __bf16* __restrict__ qkv, __bf16* __restrict__ Q,
    __bf16* __restrict__ Kd, __bf16* __restrict__ Vt)
{
    const size_t idx = (size_t)blockIdx.x * 256 + threadIdx.x;
    const int n = (int)(idx % 6144);
    const size_t mrow = idx / 6144;
    const int b = (int)(mrow >> 11), s = (int)(mrow & 2047);
    const int sel = n >> 11;           // 0=Q 1=K 2=V
    const int e = n & 2047;
    const int h = e >> 7, d = e & 127;
    const __bf16 val = qkv[idx];
    const size_t bh = (size_t)(b * HEADS + h);
    if (sel == 0)      Q [(bh * SEQ + s) * HDIM + d] = val;
    else if (sel == 1) Kd[(bh * SEQ + s) * HDIM + d] = val;
    else               Vt[(bh * HDIM + d) * SEQ + s] = val;
}

// ---------------------------------------------------------------------------
// Flash attention: block = 4 waves x 16 query rows = 64-row q tile per (b,h).
// All waves share async double-buffered K/V tiles in LDS (one fetch per block
// instead of one per wave); uniform kv loop with causal masking.
// ---------------------------------------------------------------------------
__global__ __launch_bounds__(128) void flash_attn(
    const __bf16* __restrict__ Q, const __bf16* __restrict__ Kd,
    const __bf16* __restrict__ Vt, float* __restrict__ ctx)
{
    __shared__ __bf16 KtS[2][32 * 128];  // K tile, s-major rows (s, d) — 2x8KB
    __shared__ __bf16 VtS[2][128 * 32];  // V tile, d-major rows (d, s) — 2x8KB
    __shared__ __bf16 pS[4][16 * 32];    // per-wave P scratch (C->A layout)

    const int t = threadIdx.x, w = t >> 5, lane = t & 31;
    const int qb0 = blockIdx.x * 64;
    const int qw0 = qb0 + w * 16;                   // this wave's first q row
    const int h = blockIdx.y, b = blockIdx.z;
    const size_t bh = (size_t)(b * HEADS + h);
    const __bf16* Qb = Q  + (bh * SEQ + qw0) * HDIM;
    const __bf16* Kb = Kd + bh * SEQ * HDIM;
    const __bf16* Vb = Vt + bh * HDIM * SEQ;

    // Q fragments for the whole head dim (4 x 16x32)
    v16bf qf[4];
    #pragma unroll
    for (int kc = 0; kc < 4; ++kc) qf[kc] = load_frag_a(Qb + kc * 32, HDIM);

    v8f o[8];
    float mi[8], li[8];
    #pragma unroll
    for (int f = 0; f < 8; ++f) o[f] = v8f{0.f,0.f,0.f,0.f,0.f,0.f,0.f,0.f};
    #pragma unroll
    for (int v = 0; v < 8; ++v) { mi[v] = -1e30f; li[v] = 0.f; }

    // base-2 softmax: fold log2(e) into the 1/sqrt(d) scale, use exp2f
    const float scale = 0.08838834764831845f * 1.4426950408889634f;
    const int cN = lane & 15, hiL = lane >> 4;
    __bf16* myP = pS[w];

    // Async staging: thread t copies 64B of K (row t/4, chunk t%4) and 64B of
    // V^T (row d = t) per tile => 8 async b128 per thread per step.
    const int krow = t >> 2, kchunk = (t & 3) * 32;
    auto prefetch = [&](int kv0, int buf) {
        const __bf16* ks = Kb + (size_t)(kv0 + krow) * HDIM + kchunk;
        const __bf16* vs = Vb + (size_t)t * SEQ + kv0;
        const unsigned kdst = (unsigned)(size_t)&KtS[buf][krow * 128 + kchunk];
        const unsigned vdst = (unsigned)(size_t)&VtS[buf][t * 32];
        #pragma unroll
        for (int c = 0; c < 4; ++c) {
            async_copy_b128(kdst + c * 16, ks + c * 8);
            async_copy_b128(vdst + c * 16, vs + c * 8);
        }
    };

    const int nsteps = (qb0 + 64) >> 5;             // uniform across the block
    prefetch(0, 0);

    for (int i = 0; i < nsteps; ++i) {
        const int kv0 = i << 5, buf = i & 1;
        __syncthreads();   // everyone done reading the buffer we overwrite next
        if (i + 1 < nsteps) {
            prefetch((i + 1) << 5, buf ^ 1);
            asm volatile("s_wait_asynccnt 0x8" ::: "memory");  // tile i landed
        } else {
            asm volatile("s_wait_asynccnt 0x0" ::: "memory");
        }
        __syncthreads();   // tile i visible to all waves

        // S = Q K^T for 32 kv columns (2 n-chunks x 4 k-chunks) from LDS
        v8f s0 = v8f{0.f,0.f,0.f,0.f,0.f,0.f,0.f,0.f};
        v8f s1 = v8f{0.f,0.f,0.f,0.f,0.f,0.f,0.f,0.f};
        #pragma unroll
        for (int kc = 0; kc < 4; ++kc) {
            v16bf b0 = load_frag_b(&KtS[buf][0  * 128 + kc * 32], 128);
            v16bf b1 = load_frag_b(&KtS[buf][16 * 128 + kc * 32], 128);
            s0 = wmma_bf16(qf[kc], b0, s0);
            s1 = wmma_bf16(qf[kc], b1, s1);
        }
        // online softmax (row m = v + 8*hiL lives across lanes of one 16-half)
        float p0[8], p1[8];
        #pragma unroll
        for (int v = 0; v < 8; ++v) {
            const int mrow = qw0 + v + 8 * hiL;
            float a0 = s0[v] * scale, a1 = s1[v] * scale;
            if (kv0 + cN > mrow)      a0 = -1e30f;   // causal mask
            if (kv0 + 16 + cN > mrow) a1 = -1e30f;
            float vm = fmaxf(a0, a1);
            #pragma unroll
            for (int msk = 8; msk >= 1; msk >>= 1)
                vm = fmaxf(vm, __shfl_xor(vm, msk, 32));
            const float mn = fmaxf(mi[v], vm);
            const float sc = exp2f(mi[v] - mn);
            mi[v] = mn;
            const float e0 = exp2f(a0 - mn), e1 = exp2f(a1 - mn);
            float sm = e0 + e1;
            #pragma unroll
            for (int msk = 8; msk >= 1; msk >>= 1)
                sm += __shfl_xor(sm, msk, 32);
            li[v] = li[v] * sc + sm;
            p0[v] = e0; p1[v] = e1;
            #pragma unroll
            for (int f = 0; f < 8; ++f) o[f][v] *= sc;
        }
        // P: C-layout -> row-major LDS -> A fragment (in-wave DS ordering)
        #pragma unroll
        for (int v = 0; v < 8; ++v) {
            const int mr = v + 8 * hiL;
            myP[mr * 32 + cN]      = (__bf16)p0[v];
            myP[mr * 32 + 16 + cN] = (__bf16)p1[v];
        }
        asm volatile("s_wait_dscnt 0x0" ::: "memory");
        const v16bf ap = load_frag_a(myP, 32);
        // O += P * V  (V tile is d-major: n-major for the B fragment)
        #pragma unroll
        for (int f = 0; f < 8; ++f) {
            v16bf bv = load_frag_b(&VtS[buf][(f * 16) * 32], 32);
            o[f] = wmma_bf16(ap, bv, o[f]);
        }
    }

    // Epilogue: ctx[b, s, h*128 + d] in f32 for the output projection GEMM
    #pragma unroll
    for (int v = 0; v < 8; ++v) {
        const float inv = 1.0f / li[v];
        const int mrow = qw0 + v + 8 * hiL;
        float* dst = ctx + ((size_t)b * SEQ + mrow) * EMB + h * HDIM;
        #pragma unroll
        for (int f = 0; f < 8; ++f)
            dst[f * 16 + cN] = o[f][v] * inv;
    }
}

// ---------------------------------------------------------------------------
// Host-side launcher
// ---------------------------------------------------------------------------
extern "C" void kernel_launch(void* const* d_in, const int* in_sizes, int n_in,
                              void* d_out, int out_size, void* d_ws, size_t ws_size,
                              hipStream_t stream) {
    const float* x     = (const float*)d_in[0];   // [2, 2048, 2048]
    const float* w_qkv = (const float*)d_in[1];   // [2048, 6144]
    const float* w_out = (const float*)d_in[2];   // [2048, 2048]
    float* out = (float*)d_out;                   // [2, 2048, 2048]

    char* ws = (char*)d_ws;
    __bf16* qkv16 = (__bf16*)(ws);                 // 4096*6144*2  = 50,331,648 B
    __bf16* Qb    = (__bf16*)(ws + 50331648);      // 16,777,216 B
    __bf16* Kb    = (__bf16*)(ws + 67108864);      // 16,777,216 B
    __bf16* Vt    = (__bf16*)(ws + 83886080);      // 16,777,216 B
    float*  ctx   = (float*)(ws + 100663296);      // 33,554,432 B (total 128 MB)

    // 1) qkv = x @ w_qkv            [4096 x 6144], K = 2048
    gemm_f32_bf16<false><<<dim3(48, 32), 256, 0, stream>>>(
        x, w_qkv, qkv16, BATCH * SEQ, 3 * EMB, EMB);

    // 2) split/transpose into Q, K, V^T
    reorder_qkv<<<(BATCH * SEQ * 3 * EMB) / 256, 256, 0, stream>>>(
        qkv16, Qb, Kb, Vt);

    // 3) causal flash attention -> ctx (f32)
    flash_attn<<<dim3(SEQ / 64, HEADS, BATCH), 128, 0, stream>>>(
        Qb, Kb, Vt, ctx);

    // 4) out = ctx @ w_out          [4096 x 2048], K = 2048
    gemm_f32_bf16<true><<<dim3(16, 32), 256, 0, stream>>>(
        ctx, w_out, out, BATCH * SEQ, EMB, EMB);
}
